// TokenwiseSSMoE_63702954934787
// MI455X (gfx1250) — matrix operand order repaired and checked by reading, not compiled, not run on hardware
//
#include <hip/hip_runtime.h>
#include <hip/hip_bf16.h>
#include <math.h>

typedef __bf16 bf16_t;
typedef __attribute__((ext_vector_type(16))) __bf16 v16bf;
typedef __attribute__((ext_vector_type(8)))  float  v8f;

#define TOKENS (64*196)   /* 12544, divisible by 128 */
#define DIM 768
#define HID 3072
#define NEXP 6

// ---------------------------------------------------------------------------
// CDNA5 async global->LDS copy (no data VGPRs, tracked by ASYNCcnt)
// ---------------------------------------------------------------------------
static __device__ __forceinline__ void asyncLoadB128(unsigned ldsByteOff,
                                                     unsigned long long gaddr) {
    asm volatile("global_load_async_to_lds_b128 %0, %1, off"
                 :: "v"(ldsByteOff), "v"(gaddr) : "memory");
}
static __device__ __forceinline__ void waitAsync() {
    asm volatile("s_wait_asynccnt 0x0" ::: "memory");
}

// ---------------------------------------------------------------------------
// fp32 [E][K][N] -> bf16 [E][N][K] transpose-convert, 64x64 tile per block.
// ---------------------------------------------------------------------------
__global__ void cvt_transpose(const float* __restrict__ src, bf16_t* __restrict__ dst,
                              int K, int N)
{
    __shared__ bf16_t tile[64 * 72];
    const int e  = blockIdx.z;
    const int k0 = blockIdx.x * 64;
    const int n0 = blockIdx.y * 64;
    const float* s = src + ((size_t)e * K + k0) * N + n0;
    bf16_t*      d = dst + ((size_t)e * N + n0) * K + k0;
    const int tid = threadIdx.x;

#pragma unroll
    for (int i = 0; i < 4; ++i) {
        const int idx = tid + i * 256;
        const int r   = idx >> 4;
        const int c4  = (idx & 15) * 4;
        float4 v = *(const float4*)(s + (size_t)r * N + c4);
        bf16_t* t = tile + r * 72 + c4;
        t[0] = (bf16_t)v.x; t[1] = (bf16_t)v.y; t[2] = (bf16_t)v.z; t[3] = (bf16_t)v.w;
    }
    __syncthreads();
#pragma unroll
    for (int i = 0; i < 2; ++i) {
        const int idx = tid + i * 256;
        const int n   = idx >> 3;
        const int kc  = (idx & 7) * 8;
        alignas(16) bf16_t tmp[8];
#pragma unroll
        for (int j = 0; j < 8; ++j) tmp[j] = tile[(kc + j) * 72 + n];
        *(uint4*)(d + (size_t)n * K + kc) = *(const uint4*)tmp;
    }
}

// ---------------------------------------------------------------------------
// LayerNorm + routers + softmax/top-2 + gate table + zero out region.
// ---------------------------------------------------------------------------
__global__ void ln_router(const float* __restrict__ x,
                          const float* __restrict__ gamma, const float* __restrict__ beta,
                          const float* __restrict__ srw,   const float* __restrict__ srb,
                          const float* __restrict__ hrw,   const float* __restrict__ hrb,
                          bf16_t* __restrict__ Xn, float* __restrict__ gates,
                          float* __restrict__ out,
                          float* __restrict__ sl, float* __restrict__ sp,
                          float* __restrict__ ti, float* __restrict__ tp,
                          float* __restrict__ hl, float* __restrict__ hp)
{
    const int t = blockIdx.x, tid = threadIdx.x;
    __shared__ float red[256];
    __shared__ float lgs[6];

    const float* xt = x + (size_t)t * DIM;
    float v0 = xt[tid], v1 = xt[tid + 256], v2 = xt[tid + 512];

    float* ot = out + (size_t)t * DIM;
    ot[tid] = 0.f; ot[tid + 256] = 0.f; ot[tid + 512] = 0.f;

    red[tid] = v0 + v1 + v2; __syncthreads();
    for (int o = 128; o > 0; o >>= 1) { if (tid < o) red[tid] += red[tid + o]; __syncthreads(); }
    float mu = red[0] * (1.0f / DIM);
    __syncthreads();

    float d0 = v0 - mu, d1 = v1 - mu, d2 = v2 - mu;
    red[tid] = d0*d0 + d1*d1 + d2*d2; __syncthreads();
    for (int o = 128; o > 0; o >>= 1) { if (tid < o) red[tid] += red[tid + o]; __syncthreads(); }
    float inv = rsqrtf(red[0] * (1.0f / DIM) + 1e-5f);
    __syncthreads();

    float y0 = d0 * inv * gamma[tid      ] + beta[tid      ];
    float y1 = d1 * inv * gamma[tid + 256] + beta[tid + 256];
    float y2 = d2 * inv * gamma[tid + 512] + beta[tid + 512];

    bf16_t* xn = Xn + (size_t)t * DIM;
    xn[tid] = (bf16_t)y0; xn[tid + 256] = (bf16_t)y1; xn[tid + 512] = (bf16_t)y2;

    float pl[6];
#pragma unroll
    for (int e = 0; e < 4; ++e)
        pl[e] = y0 * srw[tid*4 + e] + y1 * srw[(tid+256)*4 + e] + y2 * srw[(tid+512)*4 + e];
#pragma unroll
    for (int e = 0; e < 2; ++e)
        pl[4+e] = y0 * hrw[tid*2 + e] + y1 * hrw[(tid+256)*2 + e] + y2 * hrw[(tid+512)*2 + e];

    for (int q = 0; q < 6; ++q) {
        red[tid] = pl[q]; __syncthreads();
        for (int o = 128; o > 0; o >>= 1) { if (tid < o) red[tid] += red[tid + o]; __syncthreads(); }
        if (tid == 0) lgs[q] = red[0];
        __syncthreads();
    }

    if (tid == 0) {
        float l[4], p[4];
        float mx = -1e30f;
        for (int e = 0; e < 4; ++e) { l[e] = lgs[e] + srb[e]; sl[t*4+e] = l[e]; mx = fmaxf(mx, l[e]); }
        float s = 0.f;
        for (int e = 0; e < 4; ++e) { p[e] = expf(l[e] - mx); s += p[e]; }
        float is = 1.0f / s;
        for (int e = 0; e < 4; ++e) { p[e] *= is; sp[t*4+e] = p[e]; }
        int i1 = 0; for (int e = 1; e < 4; ++e) if (p[e] > p[i1]) i1 = e;
        int i2 = (i1 == 0) ? 1 : 0;
        for (int e = 0; e < 4; ++e) if (e != i1 && p[e] > p[i2]) i2 = e;
        ti[t*2] = (float)i1; ti[t*2+1] = (float)i2;
        tp[t*2] = p[i1];     tp[t*2+1] = p[i2];
        for (int e = 0; e < 4; ++e)
            gates[t*6 + e] = (e == i1) ? p[i1] : ((e == i2) ? p[i2] : 0.0f);

        float l0 = lgs[4] + hrb[0], l1 = lgs[5] + hrb[1];
        hl[t*2] = l0; hl[t*2+1] = l1;
        float m2 = fmaxf(l0, l1);
        float q0 = expf(l0 - m2), q1 = expf(l1 - m2), qs = q0 + q1;
        q0 /= qs; q1 /= qs;
        hp[t*2] = q0; hp[t*2+1] = q1;
        gates[t*6 + 4] = q0; gates[t*6 + 5] = q1;
    }
}

// ---------------------------------------------------------------------------
// WMMA GEMM, block 256 threads (8 waves), tile 128x128, K-stage 64, LDS
// DOUBLE-BUFFERED with async global->LDS DMA (no staging VGPRs, no spills).
// Single __shared__ array => LDS byte offsets start at 0.
//   buffer b: As at b*36864, Bs at b*36864 + 18432 (bytes); row stride 144 B.
// MODE 0: H = gelu(A*W1 + b1) bf16.   MODE 1: Out += gate*(A*W2 + b2).
// ---------------------------------------------------------------------------
#define TILE_E   (128 * 72)          /* elements per tile  (18432 B) */
#define BUF_B    (2 * TILE_E * 2)    /* bytes per double-buffer step: 73728 total */
#define STAGE_B  (unsigned)(2 * TILE_E * 2 / 2) /* 18432: byte step per buffer index for prefetch dst */

template<int KDIM, int NDIM, int MODE>
__launch_bounds__(256)
__global__ void ffn_gemm(const bf16_t* __restrict__ A,
                         const bf16_t* __restrict__ Bt,   // [N][K]
                         const float*  __restrict__ bias,
                         bf16_t* __restrict__ Hout,
                         float*  __restrict__ Out,
                         const float* __restrict__ gates,
                         int eIdx)
{
    __shared__ bf16_t smem[4 * TILE_E];   // [buf0: As,Bs][buf1: As,Bs]

    const int tid  = threadIdx.x;
    const int m0   = blockIdx.y * 128;
    const int n0   = blockIdx.x * 128;
    const int lane = tid & 31;
    const int wave = tid >> 5;
    const int wm   = (wave & 3) * 32;
    const int wn   = (wave >> 2) * 64;
    const int rowA  = wm + (lane & 15);
    const int halfA = (lane >> 4) * 8;          // ISA A-layout K half select
    const int colB  = wn + (lane & 15);
    const int kbB   = (lane >> 4) * 8;          // ISA B-layout K half select

    v8f acc[2][4];
    const v8f z = {0.f,0.f,0.f,0.f,0.f,0.f,0.f,0.f};
#pragma unroll
    for (int mt = 0; mt < 2; ++mt)
#pragma unroll
        for (int nt = 0; nt < 4; ++nt) acc[mt][nt] = z;

    // staging map: thread covers rows sR+32i (i<4), k-chunk sC..sC+7
    const int sR = tid >> 3;
    const int sC = (tid & 7) * 8;

    unsigned long long ga[4], gb[4];
    unsigned la[4], lb[4];
#pragma unroll
    for (int i = 0; i < 4; ++i) {
        const int r = sR + 32 * i;
        ga[i] = (unsigned long long)(A  + (size_t)(m0 + r) * KDIM + sC);
        gb[i] = (unsigned long long)(Bt + (size_t)(n0 + r) * KDIM + sC);
        la[i] = (unsigned)(r * 72 + sC) * 2u;                 // As byte offset
        lb[i] = (unsigned)(TILE_E + r * 72 + sC) * 2u;        // Bs byte offset
    }

    // prologue: stage 0 -> buffer 0
#pragma unroll
    for (int i = 0; i < 4; ++i) { asyncLoadB128(la[i], ga[i]); asyncLoadB128(lb[i], gb[i]); }

    unsigned buf = 0;
    for (int kt = 0; kt < KDIM; kt += 64) {
        waitAsync();                 // this wave's loads into buf complete
        __syncthreads();             // all waves' loads complete

        if (kt + 64 < KDIM) {        // prefetch next stage into other buffer
            const unsigned dst = (buf ^ 1u) * STAGE_B;
#pragma unroll
            for (int i = 0; i < 4; ++i) {
                ga[i] += 128; gb[i] += 128;   // advance K by 64 elems (128 B)
                asyncLoadB128(la[i] + dst, ga[i]);
                asyncLoadB128(lb[i] + dst, gb[i]);
            }
        }

        const bf16_t* Asb = smem + buf * (2 * TILE_E);
        const bf16_t* Bsb = Asb + TILE_E;
#pragma unroll
        for (int kk = 0; kk < 2; ++kk) {
            v16bf af[2], bfr[4];
#pragma unroll
            for (int mt = 0; mt < 2; ++mt) {
                const bf16_t* p = Asb + (rowA + mt * 16) * 72 + kk * 32 + halfA;
                ((uint4*)&af[mt])[0] = *(const uint4*)(p);       // K: half+0..7
                ((uint4*)&af[mt])[1] = *(const uint4*)(p + 16);  // K: half+16..23
            }
#pragma unroll
            for (int nt = 0; nt < 4; ++nt) {
                const bf16_t* p = Bsb + (colB + nt * 16) * 72 + kk * 32 + kbB;
                ((uint4*)&bfr[nt])[0] = *(const uint4*)(p);      // K: kb+0..7
                ((uint4*)&bfr[nt])[1] = *(const uint4*)(p + 16); // K: kb+16..23
            }
#pragma unroll
            for (int mt = 0; mt < 2; ++mt)
#pragma unroll
                for (int nt = 0; nt < 4; ++nt)
                    acc[mt][nt] = __builtin_amdgcn_wmma_f32_16x16x32_bf16(
                        false, af[mt], false, bfr[nt], (short)0, acc[mt][nt], false, false);
        }
        __syncthreads();             // buf may be overwritten next iteration
        buf ^= 1u;
    }

    // C/D layout: VGPR r -> M = r + 8*(lane>=16), N = lane&15
    const int mBase = m0 + wm + ((lane >> 4) * 8);
    if (MODE == 0) {
#pragma unroll
        for (int nt = 0; nt < 4; ++nt) {
            const int n = n0 + colB + nt * 16;
            const float bn = bias[n];
#pragma unroll
            for (int mt = 0; mt < 2; ++mt)
#pragma unroll
                for (int r = 0; r < 8; ++r) {
                    const int m = mBase + mt * 16 + r;
                    float v = acc[mt][nt][r] + bn;
                    v = 0.5f * v * (1.0f + erff(v * 0.70710678118654752f)); // exact GELU
                    Hout[(size_t)m * NDIM + n] = (bf16_t)v;
                }
        }
    } else {
#pragma unroll
        for (int mt = 0; mt < 2; ++mt)
#pragma unroll
            for (int r = 0; r < 8; ++r) {
                const int m = mBase + mt * 16 + r;
                const float g = gates[m * 6 + eIdx];
#pragma unroll
                for (int nt = 0; nt < 4; ++nt) {
                    const int n = n0 + colB + nt * 16;
                    Out[(size_t)m * DIM + n] += g * (acc[mt][nt][r] + bias[n]);
                }
            }
    }
}

// ---------------------------------------------------------------------------
extern "C" void kernel_launch(void* const* d_in, const int* in_sizes, int n_in,
                              void* d_out, int out_size, void* d_ws, size_t ws_size,
                              hipStream_t stream)
{
    const float* x       = (const float*)d_in[0];
    const float* gamma   = (const float*)d_in[1];
    const float* beta    = (const float*)d_in[2];
    const float* srw     = (const float*)d_in[3];
    const float* srb     = (const float*)d_in[4];
    const float* spec_w1 = (const float*)d_in[5];
    const float* spec_b1 = (const float*)d_in[6];
    const float* spec_w2 = (const float*)d_in[7];
    const float* spec_b2 = (const float*)d_in[8];
    const float* hrw     = (const float*)d_in[9];
    const float* hrb     = (const float*)d_in[10];
    const float* sh_w1   = (const float*)d_in[11];
    const float* sh_b1   = (const float*)d_in[12];
    const float* sh_w2   = (const float*)d_in[13];
    const float* sh_b2   = (const float*)d_in[14];

    const size_t T = TOKENS;

    bf16_t* Xn    = (bf16_t*)d_ws;                       // T*DIM
    bf16_t* Hbuf  = Xn + T * DIM;                        // T*HID
    bf16_t* W1t   = Hbuf + T * (size_t)HID;              // NEXP * [HID][DIM]
    bf16_t* W2t   = W1t + (size_t)NEXP * DIM * HID;      // NEXP * [DIM][HID]
    float*  gates = (float*)(W2t + (size_t)NEXP * HID * DIM); // T*6

    float* out = (float*)d_out;          // T*DIM
    float* sl  = out + T * DIM;          // spec_logits T*4
    float* sp  = sl + T * 4;             // spec_probs  T*4
    float* ti  = sp + T * 4;             // topk_idx    T*2
    float* tp  = ti + T * 2;             // topk_probs  T*2
    float* hl  = tp + T * 2;             // sh_logits   T*2
    float* hp  = hl + T * 2;             // sh_probs    T*2

    const size_t expElems = (size_t)DIM * HID;
    cvt_transpose<<<dim3(DIM/64, HID/64, 4), 256, 0, stream>>>(spec_w1, W1t,              DIM, HID);
    cvt_transpose<<<dim3(DIM/64, HID/64, 2), 256, 0, stream>>>(sh_w1,   W1t + 4*expElems, DIM, HID);
    cvt_transpose<<<dim3(HID/64, DIM/64, 4), 256, 0, stream>>>(spec_w2, W2t,              HID, DIM);
    cvt_transpose<<<dim3(HID/64, DIM/64, 2), 256, 0, stream>>>(sh_w2,   W2t + 4*expElems, HID, DIM);

    ln_router<<<dim3((unsigned)T), 256, 0, stream>>>(x, gamma, beta, srw, srb, hrw, hrb,
                                                     Xn, gates, out, sl, sp, ti, tp, hl, hp);

    for (int e = 0; e < NEXP; ++e) {
        const bf16_t* w1e = W1t + (size_t)e * expElems;   // N=HID, K=DIM
        const bf16_t* w2e = W2t + (size_t)e * expElems;   // N=DIM, K=HID
        const float*  b1e = (e < 4) ? (spec_b1 + (size_t)e * HID) : (sh_b1 + (size_t)(e - 4) * HID);
        const float*  b2e = (e < 4) ? (spec_b2 + (size_t)e * DIM) : (sh_b2 + (size_t)(e - 4) * DIM);

        ffn_gemm<DIM, HID, 0><<<dim3(HID/128, (unsigned)(T/128)), 256, 0, stream>>>(
            Xn, w1e, b1e, Hbuf, nullptr, nullptr, e);
        ffn_gemm<HID, DIM, 1><<<dim3(DIM/128, (unsigned)(T/128)), 256, 0, stream>>>(
            Hbuf, w2e, b2e, nullptr, out, gates, e);
    }
}
